// VectorQuantizerEMA_49108656062796
// MI455X (gfx1250) — matrix operand order repaired and checked by reading, not compiled, not run on hardware
//
#include <hip/hip_runtime.h>

// ---------------------------------------------------------------------------
// VQ-VAE EMA vector quantizer for MI455X (gfx1250, wave32, WMMA)
//   N=262144 tokens, K=1024 codes, D=64 dims
// Distance matmul via v_wmma_f32_16x16x32_bf16 with 2-term bf16 split
// (hi+lo) for ~fp32 accuracy: x*c ~= hi*hi + hi*lo + lo*hi.
// ---------------------------------------------------------------------------

#define NTOK 262144
#define KCB  1024
#define DDIM 64
#define DECAY 0.99f
#define ONE_MINUS_DECAY 0.01f
#define EPSV 1e-5f
#define COMMIT 0.25f

typedef __attribute__((ext_vector_type(16))) __bf16 v16bf;
typedef __attribute__((ext_vector_type(8)))  float  v8f;
typedef __attribute__((ext_vector_type(4)))  float  vf4;
typedef __attribute__((ext_vector_type(2)))  float  vf2;

union BfVec {
  v16bf v;
  unsigned short u[16];
  vf4 f4[2];
};

__device__ __forceinline__ unsigned short f2bf_rne(float f) {
  unsigned u = __float_as_uint(f);
  u += 0x7FFFu + ((u >> 16) & 1u);
  return (unsigned short)(u >> 16);
}
__device__ __forceinline__ float bf2f(unsigned short h) {
  return __uint_as_float(((unsigned)h) << 16);
}

// ----------------------------- prep kernels --------------------------------

__global__ void vq_zero_ws(float* __restrict__ p, int n) {
  int i = blockIdx.x * blockDim.x + threadIdx.x;
  if (i < n) p[i] = 0.0f;
}

// Split fp32 codebook into bf16 hi/lo planes (stored as ushort bit patterns).
__global__ void vq_split_codebook(const float* __restrict__ cb,
                                  unsigned short* __restrict__ cbhi,
                                  unsigned short* __restrict__ cblo) {
  int i = blockIdx.x * blockDim.x + threadIdx.x;
  if (i < KCB * DDIM) {
    float e = cb[i];
    unsigned short h = f2bf_rne(e);
    cbhi[i] = h;
    cblo[i] = f2bf_rne(e - bf2f(h));
  }
}

__global__ void vq_cnorm(const float* __restrict__ cb, float* __restrict__ cnorm) {
  int k = blockIdx.x * blockDim.x + threadIdx.x;
  if (k < KCB) {
    float s = 0.0f;
    #pragma unroll
    for (int d = 0; d < DDIM; ++d) {
      float v = cb[k * DDIM + d];
      s += v * v;
    }
    cnorm[k] = s;
  }
}

// ------------------------------ main kernel --------------------------------
// Block = 256 threads = 8 waves. Each wave owns one 16-row tile of X.
// Wave loops over 64 column tiles (16 codes each) of the codebook,
// computing S = X_tile(16x64) * C_tile^T(64x16) with 6 bf16 WMMAs,
// tracking a running (minval,minidx) per (row, col%16) in registers.

__global__ __launch_bounds__(256) void
vq_main(const float* __restrict__ x,
        const float* __restrict__ cb,
        const unsigned short* __restrict__ cbhi,
        const unsigned short* __restrict__ cblo,
        const float* __restrict__ cnorm,
        float* __restrict__ counts,
        float* __restrict__ dw,
        float* __restrict__ lossAcc,
        float* __restrict__ outq,
        float* __restrict__ outidx) {
  const int lane = threadIdx.x & 31;
  const int wave = threadIdx.x >> 5;
  const int tile = blockIdx.x * 8 + wave;
  const int half = lane >> 4;   // 0 or 1
  const int res  = lane & 15;   // row (A) / col (B,C) residue
  const int rowbase = tile * 16;

  // ---- Load A tile (16 rows of X) and split into bf16 hi/lo, WMMA layout.
  // A 16x32 bf16 operand, k order used here: lanes hold M=lane%16; half 0: K = kk+[0..15] (V0-7),
  // half 1: K = kk+16+[0..15] (V0-7).
  BfVec Ahi[2], Alo[2];
  {
    const float* xr = x + (size_t)(rowbase + res) * DDIM;
    #pragma unroll
    for (int s = 0; s < 2; ++s) {
      const int kk = 32 * s;
      const int c0 = kk + half * 16;
      const int c1 = kk + half * 16 + 8;
      alignas(16) float e[16];
      *(vf4*)&e[0]  = *(const vf4*)(xr + c0);
      *(vf4*)&e[4]  = *(const vf4*)(xr + c0 + 4);
      *(vf4*)&e[8]  = *(const vf4*)(xr + c1);
      *(vf4*)&e[12] = *(const vf4*)(xr + c1 + 4);
      #pragma unroll
      for (int i = 0; i < 16; ++i) {
        unsigned short h = f2bf_rne(e[i]);
        Ahi[s].u[i] = h;
        Alo[s].u[i] = f2bf_rne(e[i] - bf2f(h));
      }
    }
  }

  // Running argmin state: C/D layout -> lane holds col n=res, rows m=half*8+j.
  float minv[8];
  int   mini[8];
  #pragma unroll
  for (int j = 0; j < 8; ++j) { minv[j] = __builtin_inff(); mini[j] = 0; }

  for (int ct = 0; ct < KCB / 16; ++ct) {
    const int n = ct * 16 + res;  // code index for this lane's column
    const unsigned short* bh = cbhi + (size_t)n * DDIM;
    const unsigned short* bl = cblo + (size_t)n * DDIM;
    v8f c = {};
    #pragma unroll
    for (int s = 0; s < 2; ++s) {
      // B 32x16 bf16 operand, k order used here (same as A): lanes 0-15 hold K=kk+[0..15] across V0-7,
      // lanes 16-31 hold K=kk+16+[0..15]  -> 16 consecutive elems per lane.
      const int kb = 32 * s + half * 16;
      BfVec Bh, Bl;
      Bh.f4[0] = *(const vf4*)(bh + kb);
      Bh.f4[1] = *(const vf4*)(bh + kb + 8);
      Bl.f4[0] = *(const vf4*)(bl + kb);
      Bl.f4[1] = *(const vf4*)(bl + kb + 8);
      c = __builtin_amdgcn_wmma_f32_16x16x32_bf16(false, Ahi[s].v, false, Bh.v,
                                                  (short)0, c, false, false);
      c = __builtin_amdgcn_wmma_f32_16x16x32_bf16(false, Ahi[s].v, false, Bl.v,
                                                  (short)0, c, false, false);
      c = __builtin_amdgcn_wmma_f32_16x16x32_bf16(false, Alo[s].v, false, Bh.v,
                                                  (short)0, c, false, false);
    }
    // dist_eff = ||c||^2 - 2 x.c  (||x||^2 constant per row -> irrelevant)
    const float cn = cnorm[n];
    #pragma unroll
    for (int j = 0; j < 8; ++j) {
      float dv = cn - 2.0f * c[j];
      if (dv < minv[j]) { minv[j] = dv; mini[j] = n; }  // n strictly increasing
    }
  }

  // ---- Cross-lane argmin within each 16-lane half (ties -> lowest index).
  #pragma unroll
  for (int j = 0; j < 8; ++j) {
    float v = minv[j];
    int ix = mini[j];
    #pragma unroll
    for (int m = 8; m >= 1; m >>= 1) {
      float ov = __shfl_xor(v, m, 16);
      int   oi = __shfl_xor(ix, m, 16);
      if (ov < v || (ov == v && oi < ix)) { v = ov; ix = oi; }
    }
    minv[j] = v;
    mini[j] = ix;
  }

  // Broadcast winning indices to all 32 lanes: rows 0-7 from lane 0,
  // rows 8-15 from lane 16.
  int iA[8], iB[8];
  #pragma unroll
  for (int j = 0; j < 8; ++j) {
    iA[j] = __shfl(mini[j], 0, 32);
    iB[j] = __shfl(mini[j], 16, 32);
  }

  // ---- Write encoding indices + per-code counts (lanes 0 and 16).
  if (res == 0) {
    #pragma unroll
    for (int j = 0; j < 8; ++j) {
      const int r = half * 8 + j;
      outidx[rowbase + r] = (float)mini[j];
      atomicAdd(&counts[mini[j]], 1.0f);
    }
  }

  // ---- Gather quantized rows from fp32 codebook, write out, accumulate
  // commitment loss and dw = segment_sum(x). 64 floats/row, 2 per lane.
  float ls = 0.0f;
  #pragma unroll
  for (int r = 0; r < 16; ++r) {
    const int ridx = (r < 8) ? iA[r] : iB[r - 8];  // compile-time r
    const int row = rowbase + r;
    const vf2 q  = *(const vf2*)(cb + (size_t)ridx * DDIM + lane * 2);
    const vf2 xv = *(const vf2*)(x + (size_t)row * DDIM + lane * 2);
    *(vf2*)(outq + (size_t)row * DDIM + lane * 2) = q;
    atomicAdd(&dw[(size_t)ridx * DDIM + lane * 2],     xv.x);
    atomicAdd(&dw[(size_t)ridx * DDIM + lane * 2 + 1], xv.y);
    const float dx = q.x - xv.x, dy = q.y - xv.y;
    ls += dx * dx + dy * dy;
  }
  #pragma unroll
  for (int m = 16; m >= 1; m >>= 1) ls += __shfl_xor(ls, m, 32);
  if (lane == 0) atomicAdd(lossAcc, ls);
}

// ----------------------------- finalize kernel -----------------------------
// One 1024-thread block: EMA updates, Laplace smoothing (block-wide sum),
// new codebook, scalar loss.

__global__ __launch_bounds__(1024) void
vq_finalize(const float* __restrict__ counts,
            const float* __restrict__ dw,
            const float* __restrict__ lossAcc,
            const float* __restrict__ ema_cs,
            const float* __restrict__ ema_w,
            float* __restrict__ out_loss,
            float* __restrict__ out_cb,
            float* __restrict__ out_ncs,
            float* __restrict__ out_emaw) {
  __shared__ float sdata[KCB];
  const int k = threadIdx.x;
  const float ncs = DECAY * ema_cs[k] + ONE_MINUS_DECAY * counts[k];
  out_ncs[k] = ncs;
  sdata[k] = ncs;
  __syncthreads();
  for (int s = KCB / 2; s > 0; s >>= 1) {
    if (k < s) sdata[k] += sdata[k + s];
    __syncthreads();
  }
  const float nsum = sdata[0];
  const float sm = (ncs + EPSV) / (nsum + (float)KCB * EPSV) * nsum;
  const float inv = 1.0f / sm;
  #pragma unroll 4
  for (int d = 0; d < DDIM; ++d) {
    const float w = DECAY * ema_w[k * DDIM + d] + ONE_MINUS_DECAY * dw[k * DDIM + d];
    out_emaw[k * DDIM + d] = w;
    out_cb[k * DDIM + d] = w * inv;
  }
  if (k == 0) out_loss[0] = COMMIT * lossAcc[0] / ((float)NTOK * (float)DDIM);
}

// ------------------------------- host launch -------------------------------

extern "C" void kernel_launch(void* const* d_in, const int* in_sizes, int n_in,
                              void* d_out, int out_size, void* d_ws, size_t ws_size,
                              hipStream_t stream) {
  const float* x      = (const float*)d_in[0];  // [N, D]
  const float* cb     = (const float*)d_in[1];  // [K, D]
  const float* ema_cs = (const float*)d_in[2];  // [K]
  const float* ema_w  = (const float*)d_in[3];  // [K, D]

  float* out = (float*)d_out;
  // Output tuple, flattened in return order:
  float* outq    = out;                                   // [N, D]
  float* outloss = out + (size_t)NTOK * DDIM;             // [1]
  float* outidx  = outloss + 1;                           // [N]
  float* outcb   = outidx + NTOK;                         // [K, D]
  float* outncs  = outcb + (size_t)KCB * DDIM;            // [K]
  float* outemaw = outncs + KCB;                          // [K, D]

  // Workspace layout (floats, then bf16 planes):
  //   [0,1024)       counts
  //   [1024,66560)   dw (K*D)
  //   [66560]        loss accumulator
  //   [66624,67648)  cnorm (K)
  //   bytes 270592.. cbhi (K*D ushort), then cblo
  float* ws_f   = (float*)d_ws;
  float* counts = ws_f;
  float* dw     = ws_f + 1024;
  float* lossA  = ws_f + 66560;
  float* cnorm  = ws_f + 66624;
  unsigned short* cbhi = (unsigned short*)((char*)d_ws + 270592);
  unsigned short* cblo = cbhi + (size_t)KCB * DDIM;

  // Zero the atomic accumulators (counts + dw + loss) -- idempotent per call.
  const int nzero = 66561;
  vq_zero_ws<<<(nzero + 255) / 256, 256, 0, stream>>>(ws_f, nzero);

  // Precompute bf16 hi/lo codebook planes + per-code squared norms.
  vq_split_codebook<<<(KCB * DDIM + 255) / 256, 256, 0, stream>>>(cb, cbhi, cblo);
  vq_cnorm<<<(KCB + 255) / 256, 256, 0, stream>>>(cb, cnorm);

  // Main WMMA distance/argmin/gather/scatter kernel.
  // 8 waves/block * 16 rows/wave = 128 rows/block -> 2048 blocks.
  vq_main<<<NTOK / 128, 256, 0, stream>>>(x, cb, cbhi, cblo, cnorm,
                                          counts, dw, lossA, outq, outidx);

  // EMA / smoothing / codebook / loss finalization.
  vq_finalize<<<1, KCB, 0, stream>>>(counts, dw, lossA, ema_cs, ema_w,
                                     outloss, outcb, outncs, outemaw);
}
